// AdaZeroEncoderLayer_49134425866963
// MI455X (gfx1250) — compile-verified, matrix operand order checked by reading, not run on hardware
//
#include <hip/hip_runtime.h>
#include <hip/hip_bf16.h>
#include <math.h>

// ---------------------------------------------------------------------------
// Types for CDNA5 (gfx1250) wave32 WMMA
// ---------------------------------------------------------------------------
typedef __attribute__((ext_vector_type(16))) __bf16 v16bf;
typedef __attribute__((ext_vector_type(8)))  float  v8f;
typedef __attribute__((ext_vector_type(4)))  unsigned int v4u;

union BFrag {
    v16bf v;
    v4u   q[2];
    unsigned short h[16];
};

__device__ __forceinline__ unsigned short f2b(float f) {
    // float -> bf16, round-to-nearest-even
    unsigned int u = __float_as_uint(f);
    unsigned int r = u + 0x7FFFu + ((u >> 16) & 1u);
    return (unsigned short)(r >> 16);
}

// ---------------------------------------------------------------------------
// Weight convert + transpose: out[n*K + k] = bf16(in[k*N + n])
// ---------------------------------------------------------------------------
__global__ __launch_bounds__(256)
void transpose_to_bf16(const float* __restrict__ in, unsigned short* __restrict__ out,
                       int K, int N) {
    size_t idx = (size_t)blockIdx.x * blockDim.x + threadIdx.x;
    size_t total = (size_t)K * N;
    if (idx >= total) return;
    int n = (int)(idx % N);
    int k = (int)(idx / N);
    out[(size_t)n * K + k] = f2b(in[idx]);
}

// ---------------------------------------------------------------------------
// AdaLN style vectors: styles[2][3][B=4][1024]
//   part 0: (1 + tanh(gamma)),  part 1: beta,  part 2: alpha
// ---------------------------------------------------------------------------
__global__ __launch_bounds__(256)
void style_kernel(const float* __restrict__ cs,
                  const float* __restrict__ w0, const float* __restrict__ b0,
                  const float* __restrict__ w1, const float* __restrict__ b1,
                  float* __restrict__ styles) {
    int idx = blockIdx.x * blockDim.x + threadIdx.x;
    if (idx >= 2 * 4 * 3072) return;
    int which = idx / (4 * 3072);
    int rem   = idx % (4 * 3072);
    int b     = rem / 3072;
    int j     = rem % 3072;
    const float* w  = which ? w1 : w0;
    const float* bb = which ? b1 : b0;
    float acc = bb[j];
#pragma unroll
    for (int i = 0; i < 16; i++) {
        float c = cs[b * 16 + i];
        float s = c / (1.0f + expf(-c));   // silu
        acc += s * w[i * 3072 + j];
    }
    int part = j / 1024, col = j % 1024;
    float v = (part == 0) ? (1.0f + tanhf(acc)) : acc;
    styles[(((size_t)which * 3 + part) * 4 + b) * 1024 + col] = v;
}

// ---------------------------------------------------------------------------
// Fused AdaLN LayerNorm -> bf16:  out = gscale*LN(x) + beta   (per row of 1024)
// ---------------------------------------------------------------------------
__global__ __launch_bounds__(256)
void adaln_norm(const float* __restrict__ x,
                const float* __restrict__ gscale, const float* __restrict__ beta,
                unsigned short* __restrict__ out) {
    __shared__ float red[256];
    const int row = blockIdx.x;          // 0..4095
    const int b   = row >> 10;
    const float* xr = x + (size_t)row * 1024;
    float v[4]; float s = 0.0f;
#pragma unroll
    for (int i = 0; i < 4; i++) { v[i] = xr[threadIdx.x + i * 256]; s += v[i]; }
    red[threadIdx.x] = s; __syncthreads();
    for (int off = 128; off > 0; off >>= 1) {
        if (threadIdx.x < off) red[threadIdx.x] += red[threadIdx.x + off];
        __syncthreads();
    }
    float mu = red[0] * (1.0f / 1024.0f);
    __syncthreads();
    float s2 = 0.0f;
#pragma unroll
    for (int i = 0; i < 4; i++) { float d = v[i] - mu; s2 += d * d; }
    red[threadIdx.x] = s2; __syncthreads();
    for (int off = 128; off > 0; off >>= 1) {
        if (threadIdx.x < off) red[threadIdx.x] += red[threadIdx.x + off];
        __syncthreads();
    }
    float rstd = rsqrtf(red[0] * (1.0f / 1024.0f) + 1e-5f);
#pragma unroll
    for (int i = 0; i < 4; i++) {
        int col = threadIdx.x + i * 256;
        float nv = gscale[b * 1024 + col] * ((v[i] - mu) * rstd) + beta[b * 1024 + col];
        out[(size_t)row * 1024 + col] = f2b(nv);
    }
}

// ---------------------------------------------------------------------------
// WMMA bf16 GEMM:  C[M,N] = A[M,K](bf16) @ Bt[N,K](bf16)^T + bias
// 8 waves/block, wave = 32x32 tile (2x2 of 16x16x32 WMMA), block = 128x64
// Epilogues: 0 = fp32 store | 1 = GELU->bf16 | 2 = res + alpha*(.) fp32
//            3 = store V transposed bf16 [B,H,64,S]
// ---------------------------------------------------------------------------
template <int MODE>
__global__ __launch_bounds__(256)
void gemm_bf16_wmma(const unsigned short* __restrict__ A,
                    const unsigned short* __restrict__ Bt,
                    const float* __restrict__ bias,
                    float* __restrict__ outF,
                    unsigned short* __restrict__ outH,
                    const float* __restrict__ res,
                    const float* __restrict__ alpha,
                    int M, int N, int K) {
    const int lane = threadIdx.x & 31;
    const int wave = threadIdx.x >> 5;
    const int wr = wave & 3, wc = wave >> 2;
    const int rowBase = blockIdx.y * 128 + wr * 32;
    const int colBase = blockIdx.x * 64 + wc * 32;
    const int half = lane >> 4;
    const int l16  = lane & 15;

    v8f acc[2][2] = {};
    for (int k0 = 0; k0 < K; k0 += 32) {
        BFrag a[2], bf[2];
#pragma unroll
        for (int t = 0; t < 2; t++) {
            const unsigned short* ap = A + (size_t)(rowBase + t * 16 + l16) * K + k0 + half * 8;
            a[t].q[0] = *(const v4u*)ap;
            a[t].q[1] = *(const v4u*)(ap + 16);
            const unsigned short* bp = Bt + (size_t)(colBase + t * 16 + l16) * K + k0 + half * 8;
            bf[t].q[0] = *(const v4u*)bp;
            bf[t].q[1] = *(const v4u*)(bp + 16);
            if (k0 + 32 < K) {
                __builtin_prefetch((const void*)(ap + 32), 0, 1);  // global_prefetch_b8
                __builtin_prefetch((const void*)(bp + 32), 0, 1);
            }
        }
#pragma unroll
        for (int i = 0; i < 2; i++)
#pragma unroll
            for (int j = 0; j < 2; j++)
                acc[i][j] = __builtin_amdgcn_wmma_f32_16x16x32_bf16(
                    false, a[i].v, false, bf[j].v, (short)0, acc[i][j], false, false);
    }
#pragma unroll
    for (int i = 0; i < 2; i++) {
#pragma unroll
        for (int j = 0; j < 2; j++) {
#pragma unroll
            for (int e = 0; e < 8; e++) {
                int m = rowBase + i * 16 + half * 8 + e;
                int n = colBase + j * 16 + l16;
                float val = acc[i][j][e] + bias[n];
                if (MODE == 0) {
                    outF[(size_t)m * N + n] = val;
                } else if (MODE == 1) {
                    float g = 0.5f * val * (1.0f + erff(val * 0.70710678118f)); // exact GELU
                    outH[(size_t)m * N + n] = f2b(g);
                } else if (MODE == 2) {
                    int bb = m >> 10;  // S = 1024
                    outF[(size_t)m * N + n] = res[(size_t)m * N + n] + alpha[bb * N + n] * val;
                } else { // MODE 3: store V as [B, H, 64, S] bf16
                    int bb = m >> 10, s = m & 1023, h = n >> 6, d = n & 63;
                    outH[(((size_t)(bb * 16 + h)) * 64 + d) * 1024 + s] = f2b(val);
                }
            }
        }
    }
}

// ---------------------------------------------------------------------------
// RoPE + convert to head-major bf16:  [B*S,1024] fp32 -> [B,H,S,64] bf16
// ---------------------------------------------------------------------------
__global__ __launch_bounds__(256)
void rope_prep(const float* __restrict__ qf, const float* __restrict__ kf,
               unsigned short* __restrict__ qb, unsigned short* __restrict__ kb) {
    int idx = blockIdx.x * blockDim.x + threadIdx.x;   // B*S*H*32 = 2097152
    if (idx >= 4 * 1024 * 16 * 32) return;
    int j = idx & 31;
    int h = (idx >> 5) & 15;
    int s = (idx >> 9) & 1023;
    int b = idx >> 19;
    float invf = expf(-(float)(2 * j) * (9.2103403720f / 64.0f)); // 10000^(-2j/64)
    float f = (float)s * invf;
    float cf = cosf(f), sf = sinf(f);
    size_t src = ((size_t)(b * 1024 + s)) * 1024 + h * 64;
    float x1q = qf[src + j], x2q = qf[src + j + 32];
    float x1k = kf[src + j], x2k = kf[src + j + 32];
    size_t dst = (((size_t)(b * 16 + h)) * 1024 + s) * 64;
    qb[dst + j]      = f2b(x1q * cf - x2q * sf);
    qb[dst + j + 32] = f2b(x2q * cf + x1q * sf);
    kb[dst + j]      = f2b(x1k * cf - x2k * sf);
    kb[dst + j + 32] = f2b(x2k * cf + x1k * sf);
}

// ---------------------------------------------------------------------------
// Flash attention, one wave per (b,h, 16-query tile).
// Q,K: [B*H, S, 64] bf16;  Vt: [B*H, 64, S] bf16;  out: [B*S, 1024] bf16
// ---------------------------------------------------------------------------
__global__ __launch_bounds__(32)
void flash_attn(const unsigned short* __restrict__ Q,
                const unsigned short* __restrict__ Kb,
                const unsigned short* __restrict__ Vt,
                const unsigned char* __restrict__ mask,
                unsigned short* __restrict__ out) {
    __shared__ unsigned short Pld[16 * 32];
    const int lane = threadIdx.x;
    const int half = lane >> 4, l16 = lane & 15;
    const int bh = blockIdx.y;
    const int b = bh >> 4, h = bh & 15;
    const int qbase = blockIdx.x * 16;
    const unsigned short* Qp = Q  + (size_t)bh * 1024 * 64;
    const unsigned short* Kp = Kb + (size_t)bh * 1024 * 64;
    const unsigned short* Vp = Vt + (size_t)bh * 64 * 1024;
    const unsigned char*  mp = mask + b * 1024;

    BFrag qa[2];
    {
        const unsigned short* qp = Qp + (size_t)(qbase + l16) * 64;
        qa[0].q[0] = *(const v4u*)(qp + half * 8);
        qa[0].q[1] = *(const v4u*)(qp + 16 + half * 8);
        qa[1].q[0] = *(const v4u*)(qp + 32 + half * 8);
        qa[1].q[1] = *(const v4u*)(qp + 48 + half * 8);
    }
    v8f acc[4] = {};
    float rmax[8], rsum[8];
#pragma unroll
    for (int i = 0; i < 8; i++) { rmax[i] = -1e30f; rsum[i] = 0.0f; }

    for (int kb0 = 0; kb0 < 1024; kb0 += 32) {
        v8f sc[2];
#pragma unroll
        for (int t = 0; t < 2; t++) {
            const unsigned short* kp = Kp + (size_t)(kb0 + t * 16 + l16) * 64;
            BFrag k0, k1;
            k0.q[0] = *(const v4u*)(kp + half * 8);
            k0.q[1] = *(const v4u*)(kp + 16 + half * 8);
            k1.q[0] = *(const v4u*)(kp + 32 + half * 8);
            k1.q[1] = *(const v4u*)(kp + 48 + half * 8);
            v8f z = {};
            z = __builtin_amdgcn_wmma_f32_16x16x32_bf16(false, qa[0].v, false, k0.v, (short)0, z, false, false);
            z = __builtin_amdgcn_wmma_f32_16x16x32_bf16(false, qa[1].v, false, k1.v, (short)0, z, false, false);
            sc[t] = z;
        }
        float mAdd0 = mp[kb0 + l16]      ? 0.0f : -1e30f;
        float mAdd1 = mp[kb0 + 16 + l16] ? 0.0f : -1e30f;
        float p0[8], p1[8], factor[8];
#pragma unroll
        for (int i = 0; i < 8; i++) {
            float s0 = sc[0][i] * 0.125f + mAdd0;
            float s1 = sc[1][i] * 0.125f + mAdd1;
            float mx = fmaxf(s0, s1);
            for (int d = 1; d < 16; d <<= 1) mx = fmaxf(mx, __shfl_xor(mx, d, 16));
            float nm = fmaxf(rmax[i], mx);
            float f0 = expf(s0 - nm), f1 = expf(s1 - nm);
            float rs = f0 + f1;
            for (int d = 1; d < 16; d <<= 1) rs += __shfl_xor(rs, d, 16);
            float fac = expf(rmax[i] - nm);
            rsum[i] = rsum[i] * fac + rs;
            rmax[i] = nm;
            factor[i] = fac;
            p0[i] = f0; p1[i] = f1;
        }
#pragma unroll
        for (int nt = 0; nt < 4; nt++)
#pragma unroll
            for (int i = 0; i < 8; i++)
                acc[nt][i] *= factor[i];
        // C-layout -> A-layout reshape of P through LDS (bf16)
#pragma unroll
        for (int i = 0; i < 8; i++) {
            int m = half * 8 + i;
            Pld[m * 32 + l16]      = f2b(p0[i]);
            Pld[m * 32 + 16 + l16] = f2b(p1[i]);
        }
        __syncthreads();
        BFrag pa;
        pa.q[0] = *(const v4u*)(&Pld[l16 * 32 + half * 8]);
        pa.q[1] = *(const v4u*)(&Pld[l16 * 32 + 16 + half * 8]);
#pragma unroll
        for (int nt = 0; nt < 4; nt++) {
            BFrag vb;
            const unsigned short* vp = Vp + (size_t)(nt * 16 + l16) * 1024 + kb0;
            vb.q[0] = *(const v4u*)(vp + half * 8);
            vb.q[1] = *(const v4u*)(vp + 16 + half * 8);
            acc[nt] = __builtin_amdgcn_wmma_f32_16x16x32_bf16(
                false, pa.v, false, vb.v, (short)0, acc[nt], false, false);
        }
        __syncthreads();
    }
#pragma unroll
    for (int i = 0; i < 8; i++) {
        int m = qbase + half * 8 + i;
        float inv = rsum[i] > 0.0f ? 1.0f / rsum[i] : 0.0f;
        size_t rowo = (size_t)(b * 1024 + m) * 1024 + h * 64;
#pragma unroll
        for (int nt = 0; nt < 4; nt++)
            out[rowo + nt * 16 + l16] = f2b(acc[nt][i] * inv);
    }
}

// ---------------------------------------------------------------------------
// Host orchestration
// ---------------------------------------------------------------------------
extern "C" void kernel_launch(void* const* d_in, const int* in_sizes, int n_in,
                              void* d_out, int out_size, void* d_ws, size_t ws_size,
                              hipStream_t stream) {
    (void)in_sizes; (void)n_in; (void)out_size; (void)ws_size;
    const float* src   = (const float*)d_in[0];
    const unsigned char* mask = (const unsigned char*)d_in[1];
    const float* cs    = (const float*)d_in[2];
    const float* a0w   = (const float*)d_in[3];
    const float* a0b   = (const float*)d_in[4];
    const float* a1w   = (const float*)d_in[5];
    const float* a1b   = (const float*)d_in[6];
    const float* wq    = (const float*)d_in[7];
    const float* bq    = (const float*)d_in[8];
    const float* wk    = (const float*)d_in[9];
    const float* bk    = (const float*)d_in[10];
    const float* wv    = (const float*)d_in[11];
    const float* bv    = (const float*)d_in[12];
    const float* wo    = (const float*)d_in[13];
    const float* bo    = (const float*)d_in[14];
    const float* w1    = (const float*)d_in[15];
    const float* b1    = (const float*)d_in[16];
    const float* w2    = (const float*)d_in[17];
    const float* b2    = (const float*)d_in[18];
    float* outp = (float*)d_out;

    // workspace carve-up
    size_t off = 0;
    auto carve = [&](size_t bytes) -> char* {
        char* p = (char*)d_ws + off;
        off = (off + bytes + 255) & ~(size_t)255;
        return p;
    };
    float*          styles = (float*)carve(6 * 4 * 1024 * sizeof(float));
    unsigned short* wqT    = (unsigned short*)carve((size_t)1024 * 1024 * 2);
    unsigned short* wkT    = (unsigned short*)carve((size_t)1024 * 1024 * 2);
    unsigned short* wvT    = (unsigned short*)carve((size_t)1024 * 1024 * 2);
    unsigned short* woT    = (unsigned short*)carve((size_t)1024 * 1024 * 2);
    unsigned short* w1T    = (unsigned short*)carve((size_t)4096 * 1024 * 2);
    unsigned short* w2T    = (unsigned short*)carve((size_t)1024 * 4096 * 2);
    unsigned short* normedb= (unsigned short*)carve((size_t)4096 * 1024 * 2);
    float*          qf     = (float*)carve((size_t)4096 * 1024 * 4);
    float*          kf     = (float*)carve((size_t)4096 * 1024 * 4);
    unsigned short* qb2    = (unsigned short*)carve((size_t)4096 * 1024 * 2);
    unsigned short* kb2    = (unsigned short*)carve((size_t)4096 * 1024 * 2);
    unsigned short* vT     = (unsigned short*)carve((size_t)4096 * 1024 * 2);
    unsigned short* attnb  = (unsigned short*)carve((size_t)4096 * 1024 * 2);
    float*          x1     = (float*)carve((size_t)4096 * 1024 * 4);
    unsigned short* hb     = (unsigned short*)carve((size_t)4096 * 4096 * 2);

    const float* gscale0 = styles;
    const float* beta0   = styles + 4096;
    const float* alpha0  = styles + 8192;
    const float* gscale1 = styles + 12288;
    const float* beta1   = styles + 16384;
    const float* alpha1  = styles + 20480;

    // 1) weights -> bf16 [N,K]
    int tpb = 256;
    transpose_to_bf16<<<(1024 * 1024 + tpb - 1) / tpb, tpb, 0, stream>>>(wq, wqT, 1024, 1024);
    transpose_to_bf16<<<(1024 * 1024 + tpb - 1) / tpb, tpb, 0, stream>>>(wk, wkT, 1024, 1024);
    transpose_to_bf16<<<(1024 * 1024 + tpb - 1) / tpb, tpb, 0, stream>>>(wv, wvT, 1024, 1024);
    transpose_to_bf16<<<(1024 * 1024 + tpb - 1) / tpb, tpb, 0, stream>>>(wo, woT, 1024, 1024);
    transpose_to_bf16<<<(4096 * 1024 + tpb - 1) / tpb, tpb, 0, stream>>>(w1, w1T, 1024, 4096);
    transpose_to_bf16<<<(4096 * 1024 + tpb - 1) / tpb, tpb, 0, stream>>>(w2, w2T, 4096, 1024);

    // 2) AdaLN style vectors
    style_kernel<<<(2 * 4 * 3072 + tpb - 1) / tpb, tpb, 0, stream>>>(cs, a0w, a0b, a1w, a1b, styles);

    // 3) LN0 -> bf16
    adaln_norm<<<4096, 256, 0, stream>>>(src, gscale0, beta0, normedb);

    // 4) Q,K,V projections (WMMA)
    dim3 g1024(1024 / 64, 4096 / 128);
    gemm_bf16_wmma<0><<<g1024, 256, 0, stream>>>(normedb, wqT, bq, qf, nullptr, nullptr, nullptr, 4096, 1024, 1024);
    gemm_bf16_wmma<0><<<g1024, 256, 0, stream>>>(normedb, wkT, bk, kf, nullptr, nullptr, nullptr, 4096, 1024, 1024);
    gemm_bf16_wmma<3><<<g1024, 256, 0, stream>>>(normedb, wvT, bv, nullptr, vT, nullptr, nullptr, 4096, 1024, 1024);

    // 5) RoPE + head-major bf16
    rope_prep<<<(4 * 1024 * 16 * 32 + tpb - 1) / tpb, tpb, 0, stream>>>(qf, kf, qb2, kb2);

    // 6) flash attention (WMMA)
    flash_attn<<<dim3(1024 / 16, 4 * 16), 32, 0, stream>>>(qb2, kb2, vT, mask, attnb);

    // 7) O-proj + residual:  x1 = src + alpha0 * (attn @ wo + bo)
    gemm_bf16_wmma<2><<<g1024, 256, 0, stream>>>(attnb, woT, bo, x1, nullptr, src, alpha0, 4096, 1024, 1024);

    // 8) LN1 -> bf16
    adaln_norm<<<4096, 256, 0, stream>>>(x1, gscale1, beta1, normedb);

    // 9) FFN up + exact GELU -> bf16
    dim3 g4096(4096 / 64, 4096 / 128);
    gemm_bf16_wmma<1><<<g4096, 256, 0, stream>>>(normedb, w1T, b1, nullptr, hb, nullptr, nullptr, 4096, 4096, 1024);

    // 10) FFN down + residual:  out = x1 + alpha1 * (h @ w2 + b2)
    gemm_bf16_wmma<2><<<g1024, 256, 0, stream>>>(hb, w2T, b2, outp, nullptr, x1, alpha1, 4096, 1024, 4096);
}
